// BindsNetModule_69612829933942
// MI455X (gfx1250) — matrix operand order repaired and checked by reading, not compiled, not run on hardware
//
#include <hip/hip_runtime.h>
#include <stdint.h>

// ---------------------------------------------------------------------------
// BindsNet LIF layer, T=512, B=32, N=4096 on MI455X (gfx1250, wave32).
//
// The recurrence does not feed the matmul, so:
//   (1) GEMM  X[16384,4096] = input @ w   (~550 GFLOP, v_wmma_f32_16x16x32_f16)
//   (2) per-neuron LIF scan over t (parallel over B*N), in place in d_out.
//
// Roofline @ 23.3 TB/s: total HBM traffic ~1.1 GB -> ~47 us floor; f16 WMMA
// math ~100 us is the binding limit, so the GEMM is software-pipelined:
// double-buffered LDS, next tile's global loads + TDM issued before the
// current tile's 16 WMMAs, cvt+ds_store afterwards, one barrier per K-slab.
// Wave tile 64x64 (16 wmma : 16 ds_load_b128 per slab).
// d_ws holds a transposed f16 copy of w (32 MB) so B tiles are contiguous
// rows staged by the Tensor Data Mover with D# LDS padding (80B stride,
// bank-conflict-free b128 fragment reads).
// ---------------------------------------------------------------------------

typedef _Float16 v16h __attribute__((ext_vector_type(16)));
typedef float    v8f  __attribute__((ext_vector_type(8)));
typedef _Float16 h4   __attribute__((ext_vector_type(4)));
typedef float    f4   __attribute__((ext_vector_type(4)));
typedef int      i4   __attribute__((ext_vector_type(4)));
typedef unsigned u4v  __attribute__((ext_vector_type(4)));
typedef int      i8v  __attribute__((ext_vector_type(8)));

#if defined(__has_builtin)
#  if __has_builtin(__builtin_amdgcn_tensor_load_to_lds) && \
      __has_builtin(__builtin_amdgcn_s_wait_tensorcnt)
#    define HAVE_TDM 1
#  endif
#endif
#ifndef HAVE_TDM
#  define HAVE_TDM 0
#endif

#define Tn 512
#define Bn 32
#define Nn 4096
#define Mn (Tn * Bn)

#define BM 256
#define BN 128
#define BK 32
#define ASTR 40  // halves: 80B row stride -> 20-bank step, conflict-free b128
#define BSTR 40

union FragH { i4 i[2]; v16h h; };

// ---- kernel 0: wT[n][k] = (f16) w[k][n] ------------------------------------
__global__ __launch_bounds__(256)
void wt_convert_kernel(const float* __restrict__ w, _Float16* __restrict__ wT) {
  __shared__ _Float16 tile[32][34];
  const int n0 = blockIdx.x * 32, k0 = blockIdx.y * 32;
  const int tx = threadIdx.x, ty = threadIdx.y;
#pragma unroll
  for (int r = 0; r < 4; ++r)
    tile[ty + 8 * r][tx] = (_Float16)w[(size_t)(k0 + ty + 8 * r) * Nn + n0 + tx];
  __syncthreads();
#pragma unroll
  for (int r = 0; r < 4; ++r)
    wT[(size_t)(n0 + ty + 8 * r) * Nn + k0 + tx] = tile[tx][ty + 8 * r];
}

// ---- kernel 1: X = input @ w, pipelined double-buffered WMMA GEMM ----------
__global__ __launch_bounds__(256)
void gemm_wmma_kernel(const float* __restrict__ Ag,
                      const _Float16* __restrict__ BTg,
                      const float* __restrict__ Wf32,
                      float* __restrict__ Xg,
                      int use_wt) {
  __shared__ _Float16 As[2][BM * ASTR];  // 2 x 20 KB
  __shared__ _Float16 Bs[2][BN * BSTR];  // 2 x 10 KB

  const int tid  = threadIdx.x;
  const int lane = tid & 31;
  const int wid  = tid >> 5;
  const int hx   = lane >> 4;   // 0: lanes 0-15, 1: lanes 16-31
  const int l16  = lane & 15;

  const int gm0 = blockIdx.y * BM;
  const int gn0 = blockIdx.x * BN;
  const int wm  = (wid >> 1) * 64;  // 4 wave rows (M)
  const int wn  = (wid & 1) * 64;   // 2 wave cols (N)

  v8f acc[4][4] = {};

  f4 fa[8];  // next A tile in flight (f32, cvt after compute)
  i4 fb[2];  // next B tile in flight (manual path)
  (void)fb;

  // -- staging helpers ------------------------------------------------------
  auto loadA = [&](int kt) {
#pragma unroll
    for (int i = 0; i < 8; ++i) {
      int id  = tid + 256 * i;  // 2048 chunks of 4 floats
      int row = id >> 3;
      int c4  = (id & 7) << 2;
      fa[i] = *(const f4*)(Ag + (size_t)(gm0 + row) * Nn + kt + c4);
    }
  };
  auto storeA = [&](int buf) {
#pragma unroll
    for (int i = 0; i < 8; ++i) {
      int id  = tid + 256 * i;
      int row = id >> 3;
      int c4  = (id & 7) << 2;
      h4 hv = {(_Float16)fa[i].x, (_Float16)fa[i].y,
               (_Float16)fa[i].z, (_Float16)fa[i].w};
      *(h4*)(&As[buf][row * ASTR + c4]) = hv;
    }
  };

#if HAVE_TDM
  const unsigned bs_lds0 = (unsigned)(unsigned long long)(const void*)&Bs[0][0];
  const unsigned bs_lds1 = (unsigned)(unsigned long long)(const void*)&Bs[1][0];
  // group1 is loop-invariant: data_size=2B, pad_enable, pad 4 DW every 16 DW
  // -> LDS row = 64B payload + 16B pad = BSTR halves
  const i8v g1c = {(1 << 16) | (1 << 20) | (3 << 22) | (3 << 25),
                   (Nn & 0xFFFF) << 16,                           // dim0 lo
                   ((Nn >> 16) & 0xFFFF) | ((Nn & 0xFFFF) << 16), // d0 hi|d1 lo
                   ((Nn >> 16) & 0xFFFF) | (BK << 16),            // d1 hi|tile0
                   BN,                                            // tile1=128
                   Nn,                                            // d0 stride lo
                   0, 0};
  auto tdmB = [&](int kt, int buf) {
    if (wid != 0) return;  // one DMA per block
    unsigned long long ga = (unsigned long long)(const void*)BTg +
                            ((unsigned long long)gn0 * Nn + kt) * 2ull;
    u4v g0 = {1u,                                    // count=1
              buf ? bs_lds1 : bs_lds0,               // LDS dest
              (unsigned)ga,                          // global_addr lo
              (unsigned)((ga >> 32) & 0x01FFFFFFu) | (2u << 30)};  // type=2
    i4 z4 = {0, 0, 0, 0};
#if __clang_major__ >= 23
    i8v z8 = {0, 0, 0, 0, 0, 0, 0, 0};
    __builtin_amdgcn_tensor_load_to_lds(g0, g1c, z4, z4, z8, 0);
#else
    __builtin_amdgcn_tensor_load_to_lds(g0, g1c, z4, z4, 0);
#endif
  };
  auto tdmWait = [&]() {
    if (wid == 0) __builtin_amdgcn_s_wait_tensorcnt(0);
  };
#else
  auto loadB = [&](int kt) {
#pragma unroll
    for (int i = 0; i < 2; ++i) {
      int id  = tid + 256 * i;  // 512 b128 chunks
      int row = id >> 2;
      int c8  = (id & 3) << 3;
      fb[i] = *(const i4*)(BTg + (size_t)(gn0 + row) * Nn + kt + c8);
    }
  };
  auto storeB = [&](int buf) {
#pragma unroll
    for (int i = 0; i < 2; ++i) {
      int id  = tid + 256 * i;
      int row = id >> 2;
      int c8  = (id & 3) << 3;
      *(i4*)(&Bs[buf][row * BSTR + c8]) = fb[i];
    }
  };
#endif

  // slow fallback if d_ws is too small for wT: transpose+convert f32 w inline
  auto stageBw32 = [&](int kt, int buf) {
#pragma unroll
    for (int i = 0; i < 16; ++i) {
      int id = tid + 256 * i;  // 4096 elements
      int kk = id >> 7;
      int nn = id & 127;
      Bs[buf][nn * BSTR + kk] = (_Float16)Wf32[(size_t)(kt + kk) * Nn + gn0 + nn];
    }
  };

  auto stageBegin = [&](int kt, int buf) {  // issue memory, no LDS writes yet
    loadA(kt);
    if (use_wt) {
#if HAVE_TDM
      tdmB(kt, buf);
#else
      loadB(kt);
#endif
    } else {
      stageBw32(kt, buf);
    }
  };
  auto stageEnd = [&](int buf) {  // drain into LDS after compute
    storeA(buf);
    if (use_wt) {
#if HAVE_TDM
      tdmWait();
#else
      storeB(buf);
#endif
    }
  };

  // -- compute: 16 WMMAs on one 32-deep K slab ------------------------------
  // A 16x32 f16: lane l16 = M row; hx selects halves {0-7,16-23}/{8-15,24-31}
  // B 32x16 f16: lane l16 = N col; hx selects halves {0-7,16-23}/{8-15,24-31}
  auto compute = [&](int buf) {
    FragH bf[4];
#pragma unroll
    for (int ni = 0; ni < 4; ++ni) {
      const _Float16* bp = &Bs[buf][(wn + 16 * ni + l16) * BSTR + hx * 8];
      bf[ni].i[0] = *(const i4*)(bp);
      bf[ni].i[1] = *(const i4*)(bp + 16);
    }
#pragma unroll
    for (int mi = 0; mi < 4; ++mi) {
      FragH af;
      const _Float16* ap = &As[buf][(wm + 16 * mi + l16) * ASTR + hx * 8];
      af.i[0] = *(const i4*)(ap);
      af.i[1] = *(const i4*)(ap + 16);
#pragma unroll
      for (int ni = 0; ni < 4; ++ni)
        acc[mi][ni] = __builtin_amdgcn_wmma_f32_16x16x32_f16(
            false, af.h, false, bf[ni].h, (short)0, acc[mi][ni], false, false);
    }
  };

  // -- pipelined main loop --------------------------------------------------
  stageBegin(0, 0);
  stageEnd(0);
  __syncthreads();

  int buf = 0;
  for (int kt = 0; kt < Nn; kt += BK) {
    const int nxt = kt + BK;
    if (nxt < Nn) stageBegin(nxt, buf ^ 1);  // global loads / TDM overlap WMMA
    compute(buf);
    if (nxt < Nn) stageEnd(buf ^ 1);
    __syncthreads();
    buf ^= 1;
  }

  // -- store C: VGPR r holds M = r + 8*hx, N = l16 --------------------------
#pragma unroll
  for (int mi = 0; mi < 4; ++mi)
#pragma unroll
    for (int ni = 0; ni < 4; ++ni) {
      size_t row0 = (size_t)(gm0 + wm + 16 * mi + hx * 8);
      int    col  = gn0 + wn + 16 * ni + l16;
#pragma unroll
      for (int r = 0; r < 8; ++r)
        __builtin_nontemporal_store(acc[mi][ni][r], Xg + (row0 + r) * Nn + col);
    }
}

// ---- kernel 2: LIF scan, in place over d_out -------------------------------
__global__ __launch_bounds__(256)
void lif_scan_kernel(float* __restrict__ X) {
  const int idx = blockIdx.x * blockDim.x + threadIdx.x;  // 0 .. B*N-1
  const float decay = 0.9900498337491681f;                // exp(-DT/TC_DECAY)
  float v  = -65.0f;                                      // REST
  float rc = 0.0f;
  float* p = X + idx;
#pragma unroll 4
  for (int t = 0; t < Tn; ++t) {
    float x = __builtin_nontemporal_load(p);
    v = __builtin_fmaf(decay, v + 65.0f, -65.0f);  // decay*(v-REST)+REST
    if (rc <= 0.0f) v += x;                        // uses pre-decrement refrac
    rc = fmaxf(rc - 1.0f, 0.0f);
    const bool s = (v >= -52.0f);                  // THRESH
    __builtin_nontemporal_store(s ? 1.0f : 0.0f, p);
    if (s) { v = -65.0f; rc = 5.0f; }              // RESET, REFRAC
    p += Bn * Nn;
  }
}

extern "C" void kernel_launch(void* const* d_in, const int* in_sizes, int n_in,
                              void* d_out, int out_size, void* d_ws, size_t ws_size,
                              hipStream_t stream) {
  (void)in_sizes; (void)n_in; (void)out_size;
  const float* input = (const float*)d_in[0];  // [T,B,N] f32
  const float* w     = (const float*)d_in[1];  // [N,N]  f32
  float*    out = (float*)d_out;               // [T,B,N] f32: X, then spikes
  _Float16* wT  = (_Float16*)d_ws;             // [N,N]  f16 transposed (32 MB)
  const int use_wt = (ws_size >= (size_t)Nn * Nn * sizeof(_Float16)) ? 1 : 0;

  if (use_wt) {
    dim3 b0(32, 8), g0(Nn / 32, Nn / 32);
    wt_convert_kernel<<<g0, b0, 0, stream>>>(w, wT);
  }
  dim3 g1(Nn / BN, Mn / BM);  // (32, 64) blocks of 256 threads (8 waves)
  gemm_wmma_kernel<<<g1, 256, 0, stream>>>(input, wT, w, out, use_wt);
  lif_scan_kernel<<<(Bn * Nn) / 256, 256, 0, stream>>>(out);
}
